// HfMistral4MoEGate_17085379904040
// MI455X (gfx1250) — compile-verified, matrix operand order checked by reading, not run on hardware
//
#include <hip/hip_runtime.h>
#include <hip/hip_bf16.h>

// MoE gate: logits = x[T,H] @ W^T[H,E] + bias; top-8 + softmax.
// T=8192, H=5120, E=64.  HBM floor: x once = 168 MB @ 23.3 TB/s ~ 7.2 us.
// Strategy:
//   pre-pass: convert+swizzle weight fp32 -> bf16 fragment layout in d_ws (655 KB, L2-resident)
//   main:     stream x fp32 (cvt_pk->bf16 in-register), stage B chunks into LDS with
//             global_load_async_to_lds_b128 (double-buffered, ASYNCcnt), consume via
//             ds_load_b128, accumulate with v_wmma_f32_16x16x32_bf16.

typedef __attribute__((ext_vector_type(16))) __bf16 v16bf;
typedef __attribute__((ext_vector_type(8)))  float  v8f;

#define H_DIM 5120
#define E_DIM 64
#define TOPK  8
#define WAVES 4
#define TOK_PER_BLOCK (WAVES * 16)

#define KCHUNK      128                       // K per staged LDS chunk (4 WMMA K-steps)
#define NCHUNK      (H_DIM / KCHUNK)          // 40
#define CHUNK_BYTES (E_DIM * KCHUNK * 2)      // 16384 B of bf16 per chunk
#define WS_B_BYTES  ((size_t)E_DIM * H_DIM * 2)  // 655360

// ---------------- pre-pass: convert + swizzle weight into fragment order ----------------
// dst pair index p = ((c*4 + t)*32 + lane)*8 + j  -> packed bf16 pair = frag elements (2j,2j+1)
// 16-bit B 32x16 fragment layout (mirror of A): lanes 0-15 hold K {0..7,16..23},
// lanes 16-31 hold K {8..15,24..31}; element pairs pack into one dword.
__global__ __launch_bounds__(256)
void prepack_weight_bf16(const float* __restrict__ w, unsigned int* __restrict__ dst) {
  const int p = blockIdx.x * blockDim.x + threadIdx.x;
  const int NP = (H_DIM / 32) * 4 * 32 * 8;
  if (p >= NP) return;
  const int j    = p & 7;
  const int lane = (p >> 3) & 31;
  const int t    = (p >> 8) & 3;
  const int c    = p >> 10;
  const int half = lane >> 4;
  const int row  = t * 16 + (lane & 15);
  const int e0   = 2 * j;
  const int koff = (e0 < 8) ? (half * 8 + e0) : (16 + half * 8 + (e0 - 8));
  const int K    = c * 32 + koff;
  union { __bf16 h[2]; unsigned int u; } pk;
  pk.h[0] = (__bf16)w[(size_t)row * H_DIM + K];
  pk.h[1] = (__bf16)w[(size_t)row * H_DIM + K + 1];
  dst[p] = pk.u;
}

// ---------------- helpers ----------------
__device__ __forceinline__ v16bf load_afrag_f32_to_bf16(const float* __restrict__ rowptr,
                                                        int k0, int half) {
  const int b0 = k0 + half * 8;
  const int b1 = k0 + 16 + half * 8;
  const float4 f0 = *(const float4*)(rowptr + b0);
  const float4 f1 = *(const float4*)(rowptr + b0 + 4);
  const float4 f2 = *(const float4*)(rowptr + b1);
  const float4 f3 = *(const float4*)(rowptr + b1 + 4);
  v16bf r;
  r[0]  = (__bf16)f0.x; r[1]  = (__bf16)f0.y; r[2]  = (__bf16)f0.z; r[3]  = (__bf16)f0.w;
  r[4]  = (__bf16)f1.x; r[5]  = (__bf16)f1.y; r[6]  = (__bf16)f1.z; r[7]  = (__bf16)f1.w;
  r[8]  = (__bf16)f2.x; r[9]  = (__bf16)f2.y; r[10] = (__bf16)f2.z; r[11] = (__bf16)f2.w;
  r[12] = (__bf16)f3.x; r[13] = (__bf16)f3.y; r[14] = (__bf16)f3.z; r[15] = (__bf16)f3.w;
  return r;
}

__device__ __forceinline__ void async_copy_b128_to_lds(unsigned int lds_byte_off,
                                                       const void* gptr) {
  asm volatile("global_load_async_to_lds_b128 %0, %1, off"
               :: "v"(lds_byte_off), "v"(gptr) : "memory");
}
__device__ __forceinline__ void wait_asynccnt_le8() {
  asm volatile("s_wait_asynccnt 0x8" ::: "memory");
}
__device__ __forceinline__ void wait_asynccnt_0() {
  asm volatile("s_wait_asynccnt 0x0" ::: "memory");
}

union BFrag { uint4 u[2]; v16bf v; };

// ---------------- shared top-k + softmax epilogue ----------------
__device__ __forceinline__ void topk_softmax_epilogue(const float* __restrict__ row,
                                                      int token, int* __restrict__ idx_out,
                                                      float* __restrict__ wt_out) {
  unsigned long long used = 0ull;
  float tv[TOPK];
  int   ti[TOPK];
#pragma unroll
  for (int kk = 0; kk < TOPK; ++kk) {
    float best = -__builtin_inff();
    int   bi   = 0;
    for (int e = 0; e < E_DIM; ++e) {
      const float v = row[e];
      if (!((used >> e) & 1ull) && v > best) { best = v; bi = e; }  // ties -> lowest idx
    }
    used |= 1ull << bi;
    tv[kk] = best; ti[kk] = bi;
  }
  const float mx = tv[0];
  float ex[TOPK], s = 0.0f;
#pragma unroll
  for (int kk = 0; kk < TOPK; ++kk) { ex[kk] = __expf(tv[kk] - mx); s += ex[kk]; }
  const float inv = 1.0f / s;
#pragma unroll
  for (int kk = 0; kk < TOPK; ++kk) {
    idx_out[(size_t)token * TOPK + kk] = ti[kk];
    wt_out [(size_t)token * TOPK + kk] = ex[kk] * inv;
  }
}

// ---------------- main kernel: async-LDS staged bf16 B ----------------
__global__ __launch_bounds__(WAVES * 32)
void moe_gate_topk_lds(const float* __restrict__ x,
                       const unsigned char* __restrict__ wsB,  // swizzled bf16 weight
                       const float* __restrict__ bias,
                       int*   __restrict__ idx_out,
                       float* __restrict__ wt_out,
                       int T) {
  __shared__ uint4 s_b[2][CHUNK_BYTES / 16];            // 2 x 16 KB double buffer
  __shared__ float s_logits[TOK_PER_BLOCK][E_DIM + 1];  // padded vs bank conflicts

  const int tid  = threadIdx.x;
  const int wave = tid >> 5;
  const int lane = tid & 31;
  const int half = lane >> 4;
  const int col  = lane & 15;

  const unsigned int lds_b_base = (unsigned int)(unsigned long long)(&s_b[0][0]);

  const int tok0 = blockIdx.x * TOK_PER_BLOCK + wave * 16;
  const float* __restrict__ arow = x + (size_t)(tok0 + col) * H_DIM;

  v8f acc0 = {}, acc1 = {}, acc2 = {}, acc3 = {};

  // stage chunk 0 into buffer 0 (each of 128 threads copies 8 x 16B)
#pragma unroll
  for (int s = 0; s < 8; ++s)
    async_copy_b128_to_lds(lds_b_base + tid * 16 + s * 2048,
                           wsB + tid * 16 + s * 2048);

  for (int c = 0; c < NCHUNK; ++c) {
    const int  cur = c & 1;
    const bool pre = (c + 1 < NCHUNK);
    if (pre) {
      const unsigned int dstoff = lds_b_base + (cur ^ 1) * CHUNK_BYTES;
      const unsigned char* src  = wsB + (size_t)(c + 1) * CHUNK_BYTES;
#pragma unroll
      for (int s = 0; s < 8; ++s)
        async_copy_b128_to_lds(dstoff + tid * 16 + s * 2048, src + tid * 16 + s * 2048);
      wait_asynccnt_le8();   // oldest 8 (chunk c) done; chunk c+1 still in flight
    } else {
      wait_asynccnt_0();
    }
    __syncthreads();         // chunk c visible to all waves

#pragma unroll
    for (int ks = 0; ks < KCHUNK / 32; ++ks) {
      const int k0 = c * KCHUNK + ks * 32;
      if (k0 + 32 < H_DIM) __builtin_prefetch(arow + k0 + 32, 0, 3);

      const v16bf a = load_afrag_f32_to_bf16(arow, k0, half);

      const uint4* bp = &s_b[cur][(size_t)((ks * 4) * 32 + lane) * 2];
      BFrag b0, b1, b2, b3;
      b0.u[0] = bp[0];         b0.u[1] = bp[1];          // tile stride = 64 uint4
      b1.u[0] = bp[64 + 0];    b1.u[1] = bp[64 + 1];
      b2.u[0] = bp[128 + 0];   b2.u[1] = bp[128 + 1];
      b3.u[0] = bp[192 + 0];   b3.u[1] = bp[192 + 1];

      acc0 = __builtin_amdgcn_wmma_f32_16x16x32_bf16(false, a, false, b0.v, (short)0, acc0, false, false);
      acc1 = __builtin_amdgcn_wmma_f32_16x16x32_bf16(false, a, false, b1.v, (short)0, acc1, false, false);
      acc2 = __builtin_amdgcn_wmma_f32_16x16x32_bf16(false, a, false, b2.v, (short)0, acc2, false, false);
      acc3 = __builtin_amdgcn_wmma_f32_16x16x32_bf16(false, a, false, b3.v, (short)0, acc3, false, false);
    }
    __syncthreads();         // reads done before this buffer is overwritten
  }

  const float bv0 = bias[0  + col];
  const float bv1 = bias[16 + col];
  const float bv2 = bias[32 + col];
  const float bv3 = bias[48 + col];
#pragma unroll
  for (int r = 0; r < 8; ++r) {
    const int m = wave * 16 + r + 8 * half;   // C layout: M = r + 8*half
    s_logits[m][0  + col] = acc0[r] + bv0;
    s_logits[m][16 + col] = acc1[r] + bv1;
    s_logits[m][32 + col] = acc2[r] + bv2;
    s_logits[m][48 + col] = acc3[r] + bv3;
  }
  __syncthreads();

  if (tid < TOK_PER_BLOCK) {
    const int token = blockIdx.x * TOK_PER_BLOCK + tid;
    if (token < T) topk_softmax_epilogue(s_logits[tid], token, idx_out, wt_out);
  }
}

// ---------------- fallback (ws too small): round-1 style, B fp32 on the fly ----------------
__global__ __launch_bounds__(WAVES * 32)
void moe_gate_topk_direct(const float* __restrict__ x, const float* __restrict__ w,
                          const float* __restrict__ bias, int* __restrict__ idx_out,
                          float* __restrict__ wt_out, int T) {
  __shared__ float s_logits[TOK_PER_BLOCK][E_DIM + 1];
  const int tid  = threadIdx.x;
  const int wave = tid >> 5;
  const int lane = tid & 31;
  const int half = lane >> 4;
  const int col  = lane & 15;
  const int tok0 = blockIdx.x * TOK_PER_BLOCK + wave * 16;

  const float* __restrict__ arow  = x + (size_t)(tok0 + col) * H_DIM;
  const float* __restrict__ brow0 = w + (size_t)(0  + col) * H_DIM;
  const float* __restrict__ brow1 = w + (size_t)(16 + col) * H_DIM;
  const float* __restrict__ brow2 = w + (size_t)(32 + col) * H_DIM;
  const float* __restrict__ brow3 = w + (size_t)(48 + col) * H_DIM;

  v8f acc0 = {}, acc1 = {}, acc2 = {}, acc3 = {};
  for (int k0 = 0; k0 < H_DIM; k0 += 32) {
    const v16bf a  = load_afrag_f32_to_bf16(arow,  k0, half);
    const v16bf b0 = load_afrag_f32_to_bf16(brow0, k0, half);
    const v16bf b1 = load_afrag_f32_to_bf16(brow1, k0, half);
    const v16bf b2 = load_afrag_f32_to_bf16(brow2, k0, half);
    const v16bf b3 = load_afrag_f32_to_bf16(brow3, k0, half);
    acc0 = __builtin_amdgcn_wmma_f32_16x16x32_bf16(false, a, false, b0, (short)0, acc0, false, false);
    acc1 = __builtin_amdgcn_wmma_f32_16x16x32_bf16(false, a, false, b1, (short)0, acc1, false, false);
    acc2 = __builtin_amdgcn_wmma_f32_16x16x32_bf16(false, a, false, b2, (short)0, acc2, false, false);
    acc3 = __builtin_amdgcn_wmma_f32_16x16x32_bf16(false, a, false, b3, (short)0, acc3, false, false);
  }
  const float bv0 = bias[0 + col], bv1 = bias[16 + col], bv2 = bias[32 + col], bv3 = bias[48 + col];
#pragma unroll
  for (int r = 0; r < 8; ++r) {
    const int m = wave * 16 + r + 8 * half;
    s_logits[m][0  + col] = acc0[r] + bv0;
    s_logits[m][16 + col] = acc1[r] + bv1;
    s_logits[m][32 + col] = acc2[r] + bv2;
    s_logits[m][48 + col] = acc3[r] + bv3;
  }
  __syncthreads();
  if (tid < TOK_PER_BLOCK) {
    const int token = blockIdx.x * TOK_PER_BLOCK + tid;
    if (token < T) topk_softmax_epilogue(s_logits[tid], token, idx_out, wt_out);
  }
}

extern "C" void kernel_launch(void* const* d_in, const int* in_sizes, int n_in,
                              void* d_out, int out_size, void* d_ws, size_t ws_size,
                              hipStream_t stream) {
  const float* x    = (const float*)d_in[0];
  const float* w    = (const float*)d_in[1];
  const float* bias = (const float*)d_in[2];

  const int T = in_sizes[0] / H_DIM;
  int*   idx_out = (int*)d_out;
  float* wt_out  = (float*)d_out + (size_t)T * TOPK;
  const int blocks = (T + TOK_PER_BLOCK - 1) / TOK_PER_BLOCK;

  if (ws_size >= WS_B_BYTES) {
    const int NP = (H_DIM / 32) * 4 * 32 * 8;           // packed pairs
    prepack_weight_bf16<<<(NP + 255) / 256, 256, 0, stream>>>(w, (unsigned int*)d_ws);
    moe_gate_topk_lds<<<blocks, WAVES * 32, 0, stream>>>(
        x, (const unsigned char*)d_ws, bias, idx_out, wt_out, T);
  } else {
    moe_gate_topk_direct<<<blocks, WAVES * 32, 0, stream>>>(x, w, bias, idx_out, wt_out, T);
  }
}
